// SwinBlock_5033701671278
// MI455X (gfx1250) — matrix operand order repaired and checked by reading, not compiled, not run on hardware
//
#include <hip/hip_runtime.h>
#include <hip/hip_bf16.h>
#include <stdint.h>

// ---------------------------------------------------------------------------
// CDNA5 (gfx1250) implementation of the complex windowed-attention block.
// All GEMM-shaped math runs through V_WMMA_F32_16X16X32_F16 (f16 in, f32 acc).
// ---------------------------------------------------------------------------

typedef _Float16 f16;
typedef __attribute__((ext_vector_type(16))) _Float16 v16h;
typedef __attribute__((ext_vector_type(8)))  _Float16 v8h;
typedef __attribute__((ext_vector_type(8)))  float    v8f;

static constexpr int kDIM   = 96;
static constexpr int kH     = 192;
static constexpr int kW     = 192;
static constexpr int kHW    = kH * kW;        // 36864
static constexpr int kHEADS = 6;
static constexpr int kNH    = 47;             // windows along H (stride 4, win 8)
static constexpr int kNW    = 95;             // windows along W (stride 2, win 4)
static constexpr int kNWIN  = kNH * kNW;      // 4465
static constexpr float kEPS = 1e-8f;

union FragH { v16h v; v8h h2[2]; f16 e[16]; };
union FragF { v8f  v; float e[8]; };

__device__ __forceinline__ v8f wmma_f16(v16h a, v16h b, v8f c) {
  return __builtin_amdgcn_wmma_f32_16x16x32_f16(false, a, false, b, (short)0, c,
                                                false, false);
}

// A fragment: 16x32 f16 tile from a row-major [M][Kstride] array.
// Lane L (g=L/16): M = mt*16 + L%16 ; halves 0..15 -> K = kt*32 + 16g + 0..15
// (contiguous, matching the B fragments).
__device__ __forceinline__ v16h load_frag_a(const f16* base, int kstride,
                                            int mt, int kt) {
  int l = threadIdx.x & 31;
  int g = l >> 4;
  int m = (mt << 4) + (l & 15);
  const f16* p = base + (size_t)m * kstride + (kt << 5) + (g << 4);
  FragH f;
  f.h2[0] = *(const v8h*)(p);
  f.h2[1] = *(const v8h*)(p + 8);
  return f.v;
}

// B fragment: 32x16 f16 tile (K x N) stored "B^T row-major": element (k,n) at
// base[n*kstride + k].  Lane L: N = nt*16 + L%16 ; halves 0..15 -> K = kt*32 +
// 16*(L/16) + 0..15 (contiguous).
__device__ __forceinline__ v16h load_frag_bt(const f16* base, int kstride,
                                             int nt, int kt) {
  int l = threadIdx.x & 31;
  int g = l >> 4;
  int n = (nt << 4) + (l & 15);
  const f16* p = base + (size_t)n * kstride + (kt << 5) + (g << 4);
  FragH f;
  f.h2[0] = *(const v8h*)(p);
  f.h2[1] = *(const v8h*)(p + 8);
  return f.v;
}

// Coverage counts from the overlapping window geometry (win 8x4, stride 4x2).
__device__ __forceinline__ float cnt_h(int h) {
  int lo = (h < 8) ? 0 : ((h - 4) >> 2);
  int hi = h >> 2; if (hi > kNH - 1) hi = kNH - 1;
  return (float)(hi - lo + 1);
}
__device__ __forceinline__ float cnt_w(int w) {
  int lo = (w < 4) ? 0 : ((w - 2) >> 1);
  int hi = w >> 1; if (hi > kNW - 1) hi = kNW - 1;
  return (float)(hi - lo + 1);
}

// ---------------------------------------------------------------------------
// Kernel 0: pack complex weight (M x K) into f16 real-GEMM form (M x 2K):
//   outr = [Wr | -Wi], outi = [Wi | Wr]
// ---------------------------------------------------------------------------
__global__ void pack_w_kernel(const float* __restrict__ wr,
                              const float* __restrict__ wi,
                              f16* __restrict__ outr, f16* __restrict__ outi,
                              int M, int K) {
  int t = blockIdx.x * blockDim.x + threadIdx.x;
  if (t >= M * K) return;
  int m = t / K, k = t - m * K;
  float a = wr[t], b = wi[t];
  size_t row = (size_t)m * (2 * K);
  outr[row + k]     = (f16)a;
  outr[row + K + k] = (f16)(-b);
  outi[row + k]     = (f16)b;
  outi[row + K + k] = (f16)a;
}

// ---------------------------------------------------------------------------
// Kernel 1: per-window fused  gather -> QKV -> attention -> proj -> fold-add.
// One block per window (4465 blocks), 256 threads (8 wave32), 60 KB LDS.
// ---------------------------------------------------------------------------
__global__ __launch_bounds__(256) void win_attn_kernel(
    const float* __restrict__ xr, const float* __restrict__ xi,
    const f16* __restrict__ qkvA_r, const f16* __restrict__ qkvA_i,
    const f16* __restrict__ projA_r, const f16* __restrict__ projA_i,
    const float* __restrict__ rel,
    float* __restrict__ foldR, float* __restrict__ foldI) {
  // 30720 halves = 61440 bytes of static LDS:
  //   [0,9216)      qkvR  [c=288][n=32]    (f16)
  //   [9216,18432)  qkvI  [c=288][n=32]
  //   [18432,30720) reg2: phase1 xw [n=32][192]; phase3+ attnbuf 6 x [32][64]
  __shared__ __align__(16) f16 smem[30720];
  f16* qkvR = smem;
  f16* qkvI = smem + 9216;
  f16* reg2 = smem + 18432;
  f16* xw   = reg2;

  const int wIdx = blockIdx.x;
  const int wy = wIdx / kNW, wx = wIdx - wy * kNW;
  const int gh0 = wy * 4, gw0 = wx * 2;
  const int tid  = threadIdx.x;
  const int lane = tid & 31;
  const int wid  = tid >> 5;
  const int g    = lane >> 4;

  // ---- Phase 1: gather window, f32 -> f16, pixel-major [n][xr(96)|xi(96)]
  for (int idx = tid; idx < 32 * 96; idx += 256) {
    int n = idx / 96, c = idx - n * 96;
    int gh = gh0 + (n >> 2), gw = gw0 + (n & 3);
    size_t ga = (size_t)c * kHW + gh * kW + gw;
    xw[n * 192 + c]      = (f16)xr[ga];
    xw[n * 192 + 96 + c] = (f16)xi[ga];
  }
  __syncthreads();

  // ---- Phase 2: QKV GEMM, M=288 N=32 K=192, real & imag parts
  for (int ts = wid; ts < 72; ts += 8) {
    int p = ts / 36, r2 = ts - p * 36;
    int mt = r2 >> 1, nt = r2 & 1;
    const f16* A = p ? qkvA_i : qkvA_r;
    v8f acc = {};
    for (int kt = 0; kt < 6; ++kt) {
      v16h a = load_frag_a(A, 192, mt, kt);
      v16h b = load_frag_bt(xw, 192, nt, kt);
      acc = wmma_f16(a, b, acc);
    }
    f16* dst = p ? qkvI : qkvR;
    int n = (nt << 4) + (lane & 15);
    FragF fc; fc.v = acc;
    #pragma unroll
    for (int r = 0; r < 8; ++r) {
      int m = (mt << 4) + r + (g << 3);
      dst[m * 32 + n] = (f16)fc.e[r];
    }
  }
  __syncthreads();

  // ---- Phase 3: attention scores  attn = q . conj(k) * SCALE + bias
  // real: [qr|qi] @ [kr;ki]   imag: [qi|-qr] @ [kr;ki]     (K = 32)
  for (int ts = wid; ts < 48; ts += 8) {
    int h = ts / 8, r3 = ts - h * 8;
    int p = r3 >> 2, mt = (r3 >> 1) & 1, nt = r3 & 1;
    // A fragment (query), scalar gather from qkv arrays
    FragH fa;
    int nq = (mt << 4) + (lane & 15);
    #pragma unroll
    for (int t = 0; t < 16; ++t) {
      int kk = (g << 4) + t;
      f16 v;
      if (p == 0) {
        v = (kk < 16) ? qkvR[(h * 16 + kk) * 32 + nq]
                      : qkvI[(h * 16 + kk - 16) * 32 + nq];
      } else {
        v = (kk < 16) ? qkvI[(h * 16 + kk) * 32 + nq]
                      : (f16)(-(float)qkvR[(h * 16 + kk - 16) * 32 + nq]);
      }
      fa.e[t] = v;
    }
    // B fragment (key)
    FragH fb;
    int mk = (nt << 4) + (lane & 15);
    #pragma unroll
    for (int t = 0; t < 16; ++t) {
      int kk = (g << 4) + t;
      fb.e[t] = (kk < 16) ? qkvR[(96 + h * 16 + kk) * 32 + mk]
                          : qkvI[(96 + h * 16 + kk - 16) * 32 + mk];
    }
    v8f acc = {};
    acc = wmma_f16(fa.v, fb.v, acc);
    // scale + relative bias (real part only), stage f16 in attn buffer
    f16* ab = reg2 + h * 2048;  // [n=32][ar(32)|ai(32)]
    FragF fc; fc.v = acc;
    int phm = mk >> 2, pwm = mk & 3;
    #pragma unroll
    for (int r = 0; r < 8; ++r) {
      int nrow = (mt << 4) + r + (g << 3);
      float val = fc.e[r] * 0.25f;  // SCALE = HD^-0.5 = 0.25
      if (p == 0) {
        int dh = (nrow >> 2) - phm + 7;
        int dw = (nrow & 3) - pwm + 3;
        val += rel[h * 105 + dh * 7 + dw];
      }
      ab[nrow * 64 + (p << 5) + mk] = (f16)val;
    }
  }
  __syncthreads();

  // ---- Phase 4: magnitude softmax per row (192 rows, one thread each)
  for (int row = tid; row < 192; row += 256) {
    int h = row / 32, n = row & 31;
    f16* pr = reg2 + h * 2048 + n * 64;
    float mx = -1e30f;
    for (int j = 0; j < 32; ++j) {
      float ar = (float)pr[j], ai = (float)pr[32 + j];
      float m = sqrtf(ar * ar + ai * ai);
      mx = fmaxf(mx, m);
    }
    float s = 0.f;
    for (int j = 0; j < 32; ++j) {
      float ar = (float)pr[j], ai = (float)pr[32 + j];
      float m = sqrtf(ar * ar + ai * ai);
      s += __expf(m - mx);
    }
    float inv = 1.f / s;
    for (int j = 0; j < 32; ++j) {
      float ar = (float)pr[j], ai = (float)pr[32 + j];
      float m = sqrtf(ar * ar + ai * ai);
      float sc = __expf(m - mx) * inv / (m + kEPS);
      pr[j]      = (f16)(ar * sc);
      pr[32 + j] = (f16)(ai * sc);
    }
  }
  __syncthreads();

  // ---- Phase 5: out = attn @ v  (K = 64 via [Ar|Ai]); write into q-slot of
  // the qkv arrays (channels 0..95 are dead now) as out[c][n].
  for (int ts = wid; ts < 24; ts += 8) {
    int h = ts / 4, r5 = ts - h * 4;
    int p = r5 >> 1, mt = r5 & 1;
    const f16* ab = reg2 + h * 2048;
    int d = lane & 15;
    int cb = (192 + h * 16 + d) * 32;  // v channel row base in qkv arrays
    v8f acc = {};
    for (int kt = 0; kt < 2; ++kt) {
      v16h a = load_frag_a(ab, 64, mt, kt);
      FragH fb;  // B: p==0 -> [Vr;-Vi], p==1 -> [Vi;Vr]
      #pragma unroll
      for (int t = 0; t < 16; ++t) {
        int j = (kt << 5) + (g << 4) + t;
        f16 v;
        if (p == 0) v = (j < 32) ? qkvR[cb + j] : (f16)(-(float)qkvI[cb + j - 32]);
        else        v = (j < 32) ? qkvI[cb + j] : qkvR[cb + j - 32];
        fb.e[t] = v;
      }
      acc = wmma_f16(a, fb.v, acc);
    }
    f16* dst = p ? qkvI : qkvR;
    FragF fc; fc.v = acc;
    #pragma unroll
    for (int r = 0; r < 8; ++r) {
      int n = (mt << 4) + r + (g << 3);
      dst[(h * 16 + d) * 32 + n] = (f16)fc.e[r];
    }
  }
  __syncthreads();

  // ---- Phase 6: proj GEMM (M=96 N=32 K=192) + fold (atomic overlap-add)
  for (int ts = wid; ts < 24; ts += 8) {
    int p = ts / 12, r6 = ts - p * 12;
    int mt = r6 >> 1, nt = r6 & 1;
    const f16* A = p ? projA_i : projA_r;
    int n = (nt << 4) + (lane & 15);
    v8f acc = {};
    for (int kt = 0; kt < 6; ++kt) {
      v16h a = load_frag_a(A, 192, mt, kt);
      FragH fb;  // B col n: [out_r(96); out_i(96)] gathered from qkv arrays
      #pragma unroll
      for (int t = 0; t < 16; ++t) {
        int kk = (kt << 5) + (g << 4) + t;
        fb.e[t] = (kk < 96) ? qkvR[kk * 32 + n] : qkvI[(kk - 96) * 32 + n];
      }
      acc = wmma_f16(a, fb.v, acc);
    }
    float* fold = p ? foldI : foldR;
    int gh = gh0 + (n >> 2), gw = gw0 + (n & 3);
    FragF fc; fc.v = acc;
    #pragma unroll
    for (int r = 0; r < 8; ++r) {
      int c = (mt << 4) + r + (g << 3);
      atomicAdd(&fold[(size_t)c * kHW + gh * kW + gw], fc.e[r]);
    }
  }
}

// ---------------------------------------------------------------------------
// Kernel 2: residual + complex BN1.  One block per (part, channel).
// Writes y1 (f32, [c][pix]) and pixel-major f16 B matrix for the MLP.
// ---------------------------------------------------------------------------
__global__ __launch_bounds__(256) void bn1_kernel(
    const float* __restrict__ xr, const float* __restrict__ xi,
    const float* __restrict__ foldR, const float* __restrict__ foldI,
    const float* __restrict__ gr, const float* __restrict__ br,
    const float* __restrict__ gi, const float* __restrict__ bi,
    float* __restrict__ y1R, float* __restrict__ y1I,
    f16* __restrict__ Bmlp) {
  int c = blockIdx.x % kDIM;
  int p = blockIdx.x / kDIM;
  const float* x    = p ? xi : xr;
  const float* fold = p ? foldI : foldR;
  const float* gg   = p ? gi : gr;
  const float* bb   = p ? bi : br;
  float* y1 = p ? y1I : y1R;

  __shared__ float red1[256], red2[256];
  float s1 = 0.f, s2 = 0.f;
  for (int pix = threadIdx.x; pix < kHW; pix += 256) {
    int hh = pix / kW, ww = pix - hh * kW;
    float cnt = cnt_h(hh) * cnt_w(ww);
    float t = x[(size_t)c * kHW + pix] + fold[(size_t)c * kHW + pix] / (cnt + kEPS);
    s1 += t; s2 += t * t;
  }
  red1[threadIdx.x] = s1; red2[threadIdx.x] = s2;
  __syncthreads();
  for (int o = 128; o > 0; o >>= 1) {
    if (threadIdx.x < o) {
      red1[threadIdx.x] += red1[threadIdx.x + o];
      red2[threadIdx.x] += red2[threadIdx.x + o];
    }
    __syncthreads();
  }
  float mu  = red1[0] / (float)kHW;
  float var = red2[0] / (float)kHW - mu * mu;
  float sc  = gg[c] * rsqrtf(var + 1e-5f);
  float sh  = bb[c] - mu * sc;
  for (int pix = threadIdx.x; pix < kHW; pix += 256) {
    int hh = pix / kW, ww = pix - hh * kW;
    float cnt = cnt_h(hh) * cnt_w(ww);
    float t = x[(size_t)c * kHW + pix] + fold[(size_t)c * kHW + pix] / (cnt + kEPS);
    float y = sc * t + sh;
    y1[(size_t)c * kHW + pix] = y;
    Bmlp[(size_t)pix * 192 + p * 96 + c] = (f16)y;
  }
}

// ---------------------------------------------------------------------------
// Kernel 3: MLP1 (M=384, K=192) + complex GELU -> hidden f16 [pix][hr|hi]
// ---------------------------------------------------------------------------
__global__ __launch_bounds__(256) void mlp1_kernel(
    const f16* __restrict__ Ar, const f16* __restrict__ Ai,
    const f16* __restrict__ Bmlp, f16* __restrict__ hidden) {
  int n0 = blockIdx.x * 32;
  int lane = threadIdx.x & 31, wid = threadIdx.x >> 5, g = lane >> 4;
  const f16* B = Bmlp + (size_t)n0 * 192;
  for (int cid = wid; cid < 48; cid += 8) {
    int mt = cid >> 1, nt = cid & 1;
    v8f ar = {}, ai = {};
    for (int kt = 0; kt < 6; ++kt) {
      v16h b = load_frag_bt(B, 192, nt, kt);
      ar = wmma_f16(load_frag_a(Ar, 192, mt, kt), b, ar);
      ai = wmma_f16(load_frag_a(Ai, 192, mt, kt), b, ai);
    }
    int np = n0 + (nt << 4) + (lane & 15);
    FragF fr, fi; fr.v = ar; fi.v = ai;
    #pragma unroll
    for (int r = 0; r < 8; ++r) {
      int m = (mt << 4) + r + (g << 3);
      float hr = fr.e[r], hi = fi.e[r];
      float mg = sqrtf(hr * hr + hi * hi);
      float s = 0.5f * (1.0f + erff(mg * 0.70710678f)) * mg / (mg + kEPS);
      hidden[(size_t)np * 768 + m]       = (f16)(hr * s);
      hidden[(size_t)np * 768 + 384 + m] = (f16)(hi * s);
    }
  }
}

// ---------------------------------------------------------------------------
// Kernel 4: MLP2 (M=96, K=768) + residual -> x2 = y1 + m  (f32)
// ---------------------------------------------------------------------------
__global__ __launch_bounds__(256) void mlp2_kernel(
    const f16* __restrict__ Ar, const f16* __restrict__ Ai,
    const f16* __restrict__ hidden,
    const float* __restrict__ y1R, const float* __restrict__ y1I,
    float* __restrict__ x2R, float* __restrict__ x2I) {
  int n0 = blockIdx.x * 32;
  int lane = threadIdx.x & 31, wid = threadIdx.x >> 5, g = lane >> 4;
  const f16* B = hidden + (size_t)n0 * 768;
  for (int cid = wid; cid < 12; cid += 8) {
    int mt = cid >> 1, nt = cid & 1;
    v8f ar = {}, ai = {};
    for (int kt = 0; kt < 24; ++kt) {
      v16h b = load_frag_bt(B, 768, nt, kt);
      ar = wmma_f16(load_frag_a(Ar, 768, mt, kt), b, ar);
      ai = wmma_f16(load_frag_a(Ai, 768, mt, kt), b, ai);
    }
    int np = n0 + (nt << 4) + (lane & 15);
    FragF fr, fi; fr.v = ar; fi.v = ai;
    #pragma unroll
    for (int r = 0; r < 8; ++r) {
      int c = (mt << 4) + r + (g << 3);
      size_t idx = (size_t)c * kHW + np;
      x2R[idx] = y1R[idx] + fr.e[r];
      x2I[idx] = y1I[idx] + fi.e[r];
    }
  }
}

// ---------------------------------------------------------------------------
// Kernel 5: complex BN2 -> final stacked output [2][96][192][192]
// ---------------------------------------------------------------------------
__global__ __launch_bounds__(256) void bn2_kernel(
    const float* __restrict__ x2R, const float* __restrict__ x2I,
    const float* __restrict__ gr, const float* __restrict__ br,
    const float* __restrict__ gi, const float* __restrict__ bi,
    float* __restrict__ out) {
  int c = blockIdx.x % kDIM;
  int p = blockIdx.x / kDIM;
  const float* x  = p ? x2I : x2R;
  const float* gg = p ? gi : gr;
  const float* bb = p ? bi : br;

  __shared__ float red1[256], red2[256];
  float s1 = 0.f, s2 = 0.f;
  for (int pix = threadIdx.x; pix < kHW; pix += 256) {
    float t = x[(size_t)c * kHW + pix];
    s1 += t; s2 += t * t;
  }
  red1[threadIdx.x] = s1; red2[threadIdx.x] = s2;
  __syncthreads();
  for (int o = 128; o > 0; o >>= 1) {
    if (threadIdx.x < o) {
      red1[threadIdx.x] += red1[threadIdx.x + o];
      red2[threadIdx.x] += red2[threadIdx.x + o];
    }
    __syncthreads();
  }
  float mu  = red1[0] / (float)kHW;
  float var = red2[0] / (float)kHW - mu * mu;
  float sc  = gg[c] * rsqrtf(var + 1e-5f);
  float sh  = bb[c] - mu * sc;
  for (int pix = threadIdx.x; pix < kHW; pix += 256) {
    float t = x[(size_t)c * kHW + pix];
    out[(size_t)p * kDIM * kHW + (size_t)c * kHW + pix] = sc * t + sh;
  }
}

// ---------------------------------------------------------------------------
extern "C" void kernel_launch(void* const* d_in, const int* in_sizes, int n_in,
                              void* d_out, int out_size, void* d_ws,
                              size_t ws_size, hipStream_t stream) {
  (void)in_sizes; (void)n_in; (void)out_size; (void)ws_size;
  const float* xr      = (const float*)d_in[0];
  const float* xi      = (const float*)d_in[1];
  const float* qkv_wr  = (const float*)d_in[2];
  const float* qkv_wi  = (const float*)d_in[3];
  const float* proj_wr = (const float*)d_in[4];
  const float* proj_wi = (const float*)d_in[5];
  const float* rel     = (const float*)d_in[6];
  const float* mlp1_wr = (const float*)d_in[7];
  const float* mlp1_wi = (const float*)d_in[8];
  const float* mlp2_wr = (const float*)d_in[9];
  const float* mlp2_wi = (const float*)d_in[10];
  const float* n1_gr = (const float*)d_in[11];
  const float* n1_br = (const float*)d_in[12];
  const float* n1_gi = (const float*)d_in[13];
  const float* n1_bi = (const float*)d_in[14];
  const float* n2_gr = (const float*)d_in[15];
  const float* n2_br = (const float*)d_in[16];
  const float* n2_gi = (const float*)d_in[17];
  const float* n2_bi = (const float*)d_in[18];

  size_t off = 0;
  char* wsb = (char*)d_ws;
  auto take = [&](size_t bytes) -> void* {
    void* p = wsb + off;
    off += (bytes + 255) & ~(size_t)255;
    return p;
  };
  f16* qkvA_r = (f16*)take((size_t)288 * 192 * 2);
  f16* qkvA_i = (f16*)take((size_t)288 * 192 * 2);
  f16* projA_r = (f16*)take((size_t)96 * 192 * 2);
  f16* projA_i = (f16*)take((size_t)96 * 192 * 2);
  f16* m1A_r = (f16*)take((size_t)384 * 192 * 2);
  f16* m1A_i = (f16*)take((size_t)384 * 192 * 2);
  f16* m2A_r = (f16*)take((size_t)96 * 768 * 2);
  f16* m2A_i = (f16*)take((size_t)96 * 768 * 2);
  float* foldR = (float*)take((size_t)kDIM * kHW * 4);
  float* foldI = (float*)take((size_t)kDIM * kHW * 4);
  float* y1R = (float*)take((size_t)kDIM * kHW * 4);
  float* y1I = (float*)take((size_t)kDIM * kHW * 4);
  float* x2R = (float*)take((size_t)kDIM * kHW * 4);
  float* x2I = (float*)take((size_t)kDIM * kHW * 4);
  f16* Bmlp = (f16*)take((size_t)kHW * 192 * 2);
  f16* hidden = (f16*)take((size_t)kHW * 768 * 2);

  hipMemsetAsync(foldR, 0, (size_t)kDIM * kHW * 4, stream);
  hipMemsetAsync(foldI, 0, (size_t)kDIM * kHW * 4, stream);

  pack_w_kernel<<<(288 * 96 + 255) / 256, 256, 0, stream>>>(
      qkv_wr, qkv_wi, qkvA_r, qkvA_i, 288, 96);
  pack_w_kernel<<<(96 * 96 + 255) / 256, 256, 0, stream>>>(
      proj_wr, proj_wi, projA_r, projA_i, 96, 96);
  pack_w_kernel<<<(384 * 96 + 255) / 256, 256, 0, stream>>>(
      mlp1_wr, mlp1_wi, m1A_r, m1A_i, 384, 96);
  pack_w_kernel<<<(96 * 384 + 255) / 256, 256, 0, stream>>>(
      mlp2_wr, mlp2_wi, m2A_r, m2A_i, 96, 384);

  win_attn_kernel<<<kNWIN, 256, 0, stream>>>(
      xr, xi, qkvA_r, qkvA_i, projA_r, projA_i, rel, foldR, foldI);

  bn1_kernel<<<2 * kDIM, 256, 0, stream>>>(
      xr, xi, foldR, foldI, n1_gr, n1_br, n1_gi, n1_bi, y1R, y1I, Bmlp);

  mlp1_kernel<<<kHW / 32, 256, 0, stream>>>(m1A_r, m1A_i, Bmlp, hidden);

  mlp2_kernel<<<kHW / 32, 256, 0, stream>>>(
      m2A_r, m2A_i, hidden, y1R, y1I, x2R, x2I);

  bn2_kernel<<<2 * kDIM, 256, 0, stream>>>(
      x2R, x2I, n2_gr, n2_br, n2_gi, n2_bi, (float*)d_out);
}
